// Net_71545565217446
// MI455X (gfx1250) — matrix operand order repaired and checked by reading, not compiled, not run on hardware
//
#include <hip/hip_runtime.h>

// Problem constants (from reference)
#define BQ 512
#define TQ 512
#define EQ 128
#define HQ 128
#define MQ 256
#define CQ 5

#define ROWS 16      // batch rows per workgroup (WMMA M granularity)
#define APAD 136     // bf16 LDS row stride (halves): 272B rows -> conflict-free b128 reads
#define MPAD 260     // fp32 LDS row stride for MLP hidden

typedef __attribute__((ext_vector_type(16))) __bf16 v16bf;
typedef __attribute__((ext_vector_type(8)))  float  v8f;

union BF16Frag { v16bf v; __bf16 h[16]; uint4 q[2]; };
union BF16x8   { __bf16 h[8]; uint4 q; };

__device__ __forceinline__ float fast_tanh(float x) {
#if __has_builtin(__builtin_amdgcn_tanhf)
  return __builtin_amdgcn_tanhf(x);      // native v_tanh_f32 (CDNA5 TRANS op)
#else
  return tanhf(x);
#endif
}

// A-matrix fragment (16x32 bf16, M x K) from a bf16 LDS tile [16][ld].
// ISA layout: lanes 0-15: M=lane, halves 0..7 = K=koff..koff+7,
// halves 8..15 = K=koff+16..koff+23; lanes 16-31: M=lane-16, koff += 8.
// Both runs are K-contiguous 16B -> two ds_load_b128, no conversion VALU.
__device__ __forceinline__ BF16Frag load_a_frag(const __bf16* base, int ld,
                                                int kstep, int lane) {
  BF16Frag r;
  const int m    = lane & 15;
  const int koff = kstep * 32 + ((lane >> 4) << 3);
  const __bf16* p = base + m * ld + koff;
  r.q[0] = *(const uint4*)(p);        // K = koff   .. koff+7
  r.q[1] = *(const uint4*)(p + 16);   // K = koff+16.. koff+23
  return r;
}

// B-matrix fragment (32x16 bf16, K x N) = transposed tile of row-major W[N][K].
// Lanes 0-15: N=lane, K=kbase..+7 then kbase+16..+23; lanes 16-31: N=lane-16, K+=8.
// One-time cost per kernel: fp32 global loads + native bf16 converts.
__device__ __forceinline__ BF16Frag load_b_frag(const float* __restrict__ W,
                                                int ldk, int nbase, int kbase,
                                                int lane) {
  BF16Frag r;
  const float* p = W + (long long)(nbase + (lane & 15)) * ldk + kbase +
                   ((lane >> 4) << 3);
#pragma unroll
  for (int i = 0; i < 16; ++i) r.h[i] = (__bf16)p[i < 8 ? i : i + 8];
  return r;
}

__global__ __launch_bounds__(256)
void rnn_fused_kernel(const int* __restrict__ x, const int* __restrict__ lengths,
                      const float* __restrict__ emb, const float* __restrict__ W_ih,
                      const float* __restrict__ b_ih, const float* __restrict__ W_hh,
                      const float* __restrict__ b_hh, const float* __restrict__ W0,
                      const float* __restrict__ b0, const float* __restrict__ W1,
                      const float* __restrict__ b1, float* __restrict__ out)
{
  __shared__ __bf16 e_buf[ROWS][APAD];   // embedding rows for current t (bf16)
  __shared__ __bf16 h_buf[ROWS][APAD];   // recurrent state h (bf16, WMMA feed)
  __shared__ float  h1_buf[ROWS][MPAD];  // MLP hidden (fp32)

  const int tid  = threadIdx.x;
  const int lane = tid & 31;
  const int wid  = tid >> 5;             // 8 waves; wave w owns N-tile [16w,16w+16)
  const int b0r  = blockIdx.x * ROWS;

  // ---- Weight B-fragments resident in VGPRs for the whole scan ----
  BF16Frag wih[4], whh[4];
#pragma unroll
  for (int ks = 0; ks < 4; ++ks) {
    wih[ks] = load_b_frag(W_ih, EQ, wid * 16, ks * 32, lane);
    whh[ks] = load_b_frag(W_hh, HQ, wid * 16, ks * 32, lane);
  }
  const int   ncol = wid * 16 + (lane & 15);
  const float bias = b_ih[ncol] + b_hh[ncol];

  // lengths for the 8 rows this lane's accumulator covers (M = 8*(lane>>4)+r)
  int lenr[8];
#pragma unroll
  for (int r = 0; r < 8; ++r) lenr[r] = lengths[b0r + ((lane >> 4) << 3) + r];

  // h = 0
  for (int i = tid; i < ROWS * APAD; i += 256) (&h_buf[0][0])[i] = (__bf16)0.f;
  float hprev[8];
#pragma unroll
  for (int r = 0; r < 8; ++r) hprev[r] = 0.f;
  __syncthreads();

  const int erow = tid >> 4;             // row this thread helps gather
  const int ecol = (tid & 15) * 8;       // 8 elements per thread (16B bf16 store)

  for (int t = 0; t < TQ; ++t) {
    // ---- P1: gather 16 embedding rows, convert once, store bf16 to LDS ----
    {
      int token = x[(b0r + erow) * TQ + t];
      const float4* src = (const float4*)(emb + (long long)token * EQ + ecol);
      float4 a = src[0];
      float4 b = src[1];
      BF16x8 pk;
      pk.h[0] = (__bf16)a.x; pk.h[1] = (__bf16)a.y;
      pk.h[2] = (__bf16)a.z; pk.h[3] = (__bf16)a.w;
      pk.h[4] = (__bf16)b.x; pk.h[5] = (__bf16)b.y;
      pk.h[6] = (__bf16)b.z; pk.h[7] = (__bf16)b.w;
      *(uint4*)&e_buf[erow][ecol] = pk.q;
      if (t + 1 < TQ) {                  // prefetch next step's gather rows
        int tok2 = x[(b0r + erow) * TQ + t + 1];
        __builtin_prefetch(emb + (long long)tok2 * EQ + ecol, 0, 1);
      }
    }
    __syncthreads();                     // e ready; prev h writes visible

    // ---- P2: two independent 4-WMMA chains (halved RAW depth) ----
    v8f acc0 = {};                       // emb_t @ W_ih^T
    v8f acc1 = {};                       // h     @ W_hh^T
#pragma unroll
    for (int ks = 0; ks < 4; ++ks) {
      BF16Frag ae = load_a_frag(&e_buf[0][0], APAD, ks, lane);
      BF16Frag ah = load_a_frag(&h_buf[0][0], APAD, ks, lane);
      acc0 = __builtin_amdgcn_wmma_f32_16x16x32_bf16(
          false, ae.v, false, wih[ks].v, (short)0, acc0, false, false);
      acc1 = __builtin_amdgcn_wmma_f32_16x16x32_bf16(
          false, ah.v, false, whh[ks].v, (short)0, acc1, false, false);
    }

    // ---- P3: tanh + right-aligned mask; carry kept in registers ----
    float hnew[8];
#pragma unroll
    for (int r = 0; r < 8; ++r) {
      float v = fast_tanh(acc0[r] + acc1[r] + bias);
      hnew[r] = (lenr[r] >= (TQ - t)) ? v : hprev[r];
    }
    __syncthreads();                     // all reads of h_buf/e_buf complete

    // ---- P4: publish h tile as bf16 (wave exclusively owns column ncol) ----
#pragma unroll
    for (int r = 0; r < 8; ++r) {
      int m = ((lane >> 4) << 3) + r;
      h_buf[m][ncol] = (__bf16)hnew[r];
      hprev[r] = hnew[r];
    }
  }
  __syncthreads();

  // ---- MLP layer 0 (WMMA): h1 = relu(h @ W0^T + b0), N = 256 ----
#pragma unroll
  for (int j = 0; j < 2; ++j) {
    const int nb = wid * 32 + j * 16;
    BF16Frag w0f[4];
#pragma unroll
    for (int ks = 0; ks < 4; ++ks) w0f[ks] = load_b_frag(W0, HQ, nb, ks * 32, lane);
    v8f acc = {};
#pragma unroll
    for (int ks = 0; ks < 4; ++ks) {
      BF16Frag a = load_a_frag(&h_buf[0][0], APAD, ks, lane);
      acc = __builtin_amdgcn_wmma_f32_16x16x32_bf16(
          false, a.v, false, w0f[ks].v, (short)0, acc, false, false);
    }
    const int   n  = nb + (lane & 15);
    const float bb = b0[n];
#pragma unroll
    for (int r = 0; r < 8; ++r) {
      int m = ((lane >> 4) << 3) + r;
      float v = acc[r] + bb;
      h1_buf[m][n] = v > 0.f ? v : 0.f;
    }
  }
  __syncthreads();

  // ---- MLP layer 1 + log_softmax (tiny: 16 rows x 5 classes) ----
  if (tid < ROWS) {
    const int row = tid;
    float z[CQ];
#pragma unroll
    for (int c = 0; c < CQ; ++c) {
      float s = b1[c];
      for (int m = 0; m < MQ; ++m) s += h1_buf[row][m] * W1[c * MQ + m];
      z[c] = s > 0.f ? s : 0.f;                    // relu before log_softmax
    }
    float mx = z[0];
#pragma unroll
    for (int c = 1; c < CQ; ++c) mx = fmaxf(mx, z[c]);
    float se = 0.f;
#pragma unroll
    for (int c = 0; c < CQ; ++c) se += __expf(z[c] - mx);
    float lse = mx + __logf(se);
#pragma unroll
    for (int c = 0; c < CQ; ++c) out[(b0r + row) * CQ + c] = z[c] - lse;
  }
}

extern "C" void kernel_launch(void* const* d_in, const int* in_sizes, int n_in,
                              void* d_out, int out_size, void* d_ws, size_t ws_size,
                              hipStream_t stream) {
  (void)in_sizes; (void)n_in; (void)d_ws; (void)ws_size; (void)out_size;
  const int*   x    = (const int*)d_in[0];
  const int*   len  = (const int*)d_in[1];
  const float* emb  = (const float*)d_in[2];
  const float* W_ih = (const float*)d_in[3];
  const float* b_ih = (const float*)d_in[4];
  const float* W_hh = (const float*)d_in[5];
  const float* b_hh = (const float*)d_in[6];
  const float* W0   = (const float*)d_in[7];
  const float* b0   = (const float*)d_in[8];
  const float* W1   = (const float*)d_in[9];
  const float* b1   = (const float*)d_in[10];
  float* out = (float*)d_out;

  dim3 grid(BQ / ROWS);   // 32 workgroups: batch-parallelism limit of the scan
  dim3 block(256);        // 8 wave32 waves per WGP
  rnn_fused_kernel<<<grid, block, 0, stream>>>(x, len, emb, W_ih, b_ih, W_hh,
                                               b_hh, W0, b0, W1, b1, out);
}
